// ModuleNet3D_86706799771688
// MI455X (gfx1250) — matrix operand order repaired and checked by reading, not compiled, not run on hardware
//
#include <hip/hip_runtime.h>
#include <math.h>

// ---------- types for WMMA ----------
typedef __attribute__((ext_vector_type(16))) __bf16 v16bf;
typedef __attribute__((ext_vector_type(8)))  float  v8f;
typedef __attribute__((ext_vector_type(4)))  float  vf4;

union FragU { vf4 q[2]; v16bf v; };

#define WMMA_BF16(A, B, C) \
  __builtin_amdgcn_wmma_f32_16x16x32_bf16(false, (A), false, (B), (short)0, (C), false, false)

// fp32 -> bf16 round-to-nearest-even (bits)
static __device__ __forceinline__ unsigned short f2bf(float f) {
  unsigned u = __float_as_uint(f);
  u += 0x7fffu + ((u >> 16) & 1u);
  return (unsigned short)(u >> 16);
}
static __device__ __forceinline__ float bf2f(unsigned short s) {
  return __uint_as_float(((unsigned)s) << 16);
}

// Problem constants (from reference setup_inputs)
#define C_IN   64
#define TIN    6
#define HIN    40
#define WIN    40
#define TOUT   12
#define HOUT   80
#define WOUT   80
#define NPOS   (TOUT*HOUT*WOUT)      // 76800
#define NPOS_IN (TIN*HIN*WIN)        // 9600
#define GC     539
#define KPAD   544                   // 17 * 32
#define KT1    17
#define KT2    8
#define MTILE  32                    // positions per block

// ws layout (bytes)
#define W1S_OFF  0
#define W1S_BYTES (KPAD*256*2)             // 278528
#define W2S_OFF  (W1S_OFF + W1S_BYTES)
#define W2S_BYTES (256*256*2)              // 131072
#define B1_OFF   (W2S_OFF + W2S_BYTES)
#define B1_BYTES (256*4)
#define SC_OFF   (B1_OFF + B1_BYTES)
#define SC_BYTES (3*NPOS_IN*4)

// ---------------- stage 0: fold pw1 (dw1 absorbed) into swizzled bf16 B-fragments ----------
// fragment order: [(ntile*17 + ktile)][lane(32)][16 consecutive K values] ; k = kt*32 + (lane>>4)*16 + i
__global__ __launch_bounds__(256) void k_fold_w1(const float* __restrict__ pw1_w,
                                                 const float* __restrict__ dw1_w,
                                                 unsigned short* __restrict__ w1s) {
  int idx = blockIdx.x * 256 + threadIdx.x;          // 0 .. 139263
  int i    = idx & 15;
  int lane = (idx >> 4) & 31;
  int rest = idx >> 9;                                // 0..271
  int kt   = rest % 17;
  int nt   = rest / 17;
  int k = kt * 32 + (lane >> 4) * 16 + i;
  int n = nt * 16 + (lane & 15);
  float v = 0.0f;
  if (k < 536) v = pw1_w[n * GC + k] * dw1_w[k];
  w1s[idx] = f2bf(v);
}

__global__ __launch_bounds__(256) void k_fold_w2(const float* __restrict__ fc1_w,
                                                 unsigned short* __restrict__ w2s) {
  int idx = blockIdx.x * 256 + threadIdx.x;          // 0 .. 65535
  int i    = idx & 15;
  int lane = (idx >> 4) & 31;
  int rest = idx >> 9;                                // 0..127
  int kt   = rest & 7;
  int nt   = rest >> 3;
  int k = kt * 32 + (lane >> 4) * 16 + i;
  int n = nt * 16 + (lane & 15);
  w2s[idx] = f2bf(fc1_w[n * 256 + k]);
}

__global__ __launch_bounds__(256) void k_b1(const float* __restrict__ pw1_w,
                                            const float* __restrict__ pw1_b,
                                            const float* __restrict__ dw1_w,
                                            const float* __restrict__ dw1_b,
                                            float* __restrict__ b1) {
  int o = threadIdx.x;  // 256
  float s = pw1_b[o];
  for (int c = 0; c < GC; ++c) s += pw1_w[o * GC + c] * dw1_b[c];
  for (int c = 536; c < GC; ++c) s += 2.0f * pw1_w[o * GC + c] * dw1_w[c];
  b1[o] = s;
}

// skip path at input resolution: sc[o][p] over 9600 positions
__global__ __launch_bounds__(256) void k_sc(const float* __restrict__ feat,
                                            const float* __restrict__ sc_dw_w,
                                            const float* __restrict__ sc_dw_b,
                                            const float* __restrict__ sc_pw_w,
                                            const float* __restrict__ sc_pw_b,
                                            float* __restrict__ sc) {
  int p = blockIdx.x * 256 + threadIdx.x;
  if (p >= NPOS_IN) return;
  float acc[3];
  for (int o = 0; o < 3; ++o) acc[o] = sc_pw_b[o];
  for (int c = 0; c < C_IN; ++c) {
    float v = feat[c * NPOS_IN + p] * sc_dw_w[c] + sc_dw_b[c];
    for (int o = 0; o < 3; ++o) acc[o] += sc_pw_w[o * C_IN + c] * v;
  }
  for (int o = 0; o < 3; ++o) sc[o * NPOS_IN + p] = acc[o];
}

// nearest-index + relative coord helper (matches jnp.round == RNE via rintf)
static __device__ __forceinline__ void nr(float co, float sgn, float n, float nm1,
                                          int& idx, float& d) {
  float c  = co + sgn * (1.0f / n) + 1e-6f;
  float cc = fminf(fmaxf(c, -1.0f + 1e-6f), 1.0f - 1e-6f);
  float x  = rintf(((cc + 1.0f) * n - 1.0f) * 0.5f);
  x = fminf(fmaxf(x, 0.0f), nm1);
  idx = (int)x;
  float l = -1.0f + (1.0f / n) + (2.0f / n) * x;
  d = (co - l) * n;
}

// ---------------- main fused kernel: grid build -> GEMM1 -> GEMM2+GELU -> fc2 + trilinear skip ----
#define GSTR 552   // grid LDS row stride (ushorts), 16B-aligned, conflict-free
#define XSTR 264   // x/h LDS row stride
#define HOFF 8448  // sH base (ushorts) inside sBuf

__global__ __launch_bounds__(256) void k_main(const float* __restrict__ feat,
                                              const unsigned short* __restrict__ w1s,
                                              const unsigned short* __restrict__ w2s,
                                              const float* __restrict__ b1,
                                              const float* __restrict__ fc1_b,
                                              const float* __restrict__ fc2_w,
                                              const float* __restrict__ fc2_b,
                                              const float* __restrict__ sc,
                                              float* __restrict__ out) {
  __shared__ unsigned short sBuf[MTILE * GSTR];  // 35328 B, reused as grid -> (x | h)
  __shared__ int   sOff[MTILE][8];
  __shared__ float sWgt[MTILE][8];

  const int tid  = threadIdx.x;
  const int pos0 = blockIdx.x * MTILE;

  // ---- per-position indices / rel coords / area weights ----
  if (tid < MTILE) {
    int p = pos0 + tid;
    int t = p / (HOUT * WOUT);
    int rem = p - t * (HOUT * WOUT);
    int h = rem / WOUT, w = rem - h * WOUT;
    float ct = -1.0f + (1.0f / TOUT) + (2.0f / TOUT) * t;
    float ch = -1.0f + (1.0f / HOUT) + (2.0f / HOUT) * h;
    float cw = -1.0f + (1.0f / WOUT) + (2.0f / WOUT) * w;
    int it[2], ih[2], iw[2]; float dct[2], dch[2], dcw[2];
    for (int s2 = 0; s2 < 2; ++s2) {
      float sg = s2 ? 1.0f : -1.0f;
      nr(ct, sg, (float)TIN, (float)(TIN - 1), it[s2], dct[s2]);
      nr(ch, sg, (float)HIN, (float)(HIN - 1), ih[s2], dch[s2]);
      nr(cw, sg, (float)WIN, (float)(WIN - 1), iw[s2], dcw[s2]);
    }
    float areas[8], tot = 0.0f;
    for (int q = 0; q < 8; ++q) {
      int a = q >> 2, b = (q >> 1) & 1, c2 = q & 1;
      areas[q] = fabsf(dct[a]) * fabsf(dch[b]) * fabsf(dcw[c2]) + 1e-9f;
      tot += areas[q];
    }
    unsigned short* grow = &sBuf[tid * GSTR];
    for (int q = 0; q < 8; ++q) {
      int a = q >> 2, b = (q >> 1) & 1, c2 = q & 1;
      sOff[tid][q] = it[a] * (HIN * WIN) + ih[b] * WIN + iw[c2];
      sWgt[tid][q] = areas[7 - q] / tot;
      grow[q * 3 + 0] = f2bf(dct[a]);
      grow[q * 3 + 1] = f2bf(dch[b]);
      grow[q * 3 + 2] = f2bf(dcw[c2]);
    }
    for (int c = 536; c < KPAD; ++c) grow[c] = 0;  // zero K padding
  }
  __syncthreads();

  // ---- gather weighted feat into grid channels 24..535 ----
  for (int i = tid; i < MTILE * 8 * C_IN; i += 256) {
    int pl = i >> 9;           // /512
    int rem = i & 511;
    int q = rem >> 6, c = rem & 63;
    float v = feat[c * NPOS_IN + sOff[pl][q]] * sWgt[pl][q];
    sBuf[pl * GSTR + 24 + q * C_IN + c] = f2bf(v);
  }
  __syncthreads();

  const int wave = tid >> 5, lane = tid & 31;
  const int lrow = lane & 15, lhi = lane >> 4;
  const int colb = wave * 32;
  const vf4* w1q = (const vf4*)w1s;
  const vf4* w2q = (const vf4*)w2s;
  const v8f vzero = {0.f, 0.f, 0.f, 0.f, 0.f, 0.f, 0.f, 0.f};

  // ---- GEMM1: x[32x256] = grid[32x544] * W1 ----
  v8f a00 = vzero, a01 = vzero, a10 = vzero, a11 = vzero;
  {
    const int nt0 = wave * 2, nt1 = nt0 + 1;
#pragma unroll
    for (int kt = 0; kt < KT1; ++kt) {
      const int k0 = kt * 32 + lhi * 16;
      FragU A0, A1, B0, B1;
      A0.q[0] = *(const vf4*)&sBuf[lrow * GSTR + k0];
      A0.q[1] = *(const vf4*)&sBuf[lrow * GSTR + k0 + 8];
      A1.q[0] = *(const vf4*)&sBuf[(16 + lrow) * GSTR + k0];
      A1.q[1] = *(const vf4*)&sBuf[(16 + lrow) * GSTR + k0 + 8];
      int f0 = ((nt0 * KT1 + kt) * 32 + lane) * 2;
      int f1 = ((nt1 * KT1 + kt) * 32 + lane) * 2;
      B0.q[0] = w1q[f0]; B0.q[1] = w1q[f0 + 1];
      B1.q[0] = w1q[f1]; B1.q[1] = w1q[f1 + 1];
      a00 = WMMA_BF16(A0.v, B0.v, a00);
      a01 = WMMA_BF16(A0.v, B1.v, a01);
      a10 = WMMA_BF16(A1.v, B0.v, a10);
      a11 = WMMA_BF16(A1.v, B1.v, a11);
    }
  }
  __syncthreads();  // all waves done reading grid before x overwrites it

  {  // bias + store x (bf16) into sBuf[0 .. 32*264)
    float b0v = b1[colb + lrow], b1v = b1[colb + 16 + lrow];
#pragma unroll
    for (int r = 0; r < 8; ++r) {
      int r0 = lhi * 8 + r, r1 = 16 + lhi * 8 + r;
      sBuf[r0 * XSTR + colb + lrow]      = f2bf(a00[r] + b0v);
      sBuf[r0 * XSTR + colb + 16 + lrow] = f2bf(a01[r] + b1v);
      sBuf[r1 * XSTR + colb + lrow]      = f2bf(a10[r] + b0v);
      sBuf[r1 * XSTR + colb + 16 + lrow] = f2bf(a11[r] + b1v);
    }
  }
  __syncthreads();

  // ---- GEMM2: h = gelu(x * W2 + fc1_b) ----
  a00 = vzero; a01 = vzero; a10 = vzero; a11 = vzero;
  {
    const int nt0 = wave * 2, nt1 = nt0 + 1;
#pragma unroll
    for (int kt = 0; kt < KT2; ++kt) {
      const int k0 = kt * 32 + lhi * 16;
      FragU A0, A1, B0, B1;
      A0.q[0] = *(const vf4*)&sBuf[lrow * XSTR + k0];
      A0.q[1] = *(const vf4*)&sBuf[lrow * XSTR + k0 + 8];
      A1.q[0] = *(const vf4*)&sBuf[(16 + lrow) * XSTR + k0];
      A1.q[1] = *(const vf4*)&sBuf[(16 + lrow) * XSTR + k0 + 8];
      int f0 = ((nt0 * KT2 + kt) * 32 + lane) * 2;
      int f1 = ((nt1 * KT2 + kt) * 32 + lane) * 2;
      B0.q[0] = w2q[f0]; B0.q[1] = w2q[f0 + 1];
      B1.q[0] = w2q[f1]; B1.q[1] = w2q[f1 + 1];
      a00 = WMMA_BF16(A0.v, B0.v, a00);
      a01 = WMMA_BF16(A0.v, B1.v, a01);
      a10 = WMMA_BF16(A1.v, B0.v, a10);
      a11 = WMMA_BF16(A1.v, B1.v, a11);
    }
  }
  {  // bias + exact gelu, store h (disjoint LDS region: no barrier needed before writes)
    float b0v = fc1_b[colb + lrow], b1v = fc1_b[colb + 16 + lrow];
#pragma unroll
    for (int r = 0; r < 8; ++r) {
      int r0 = lhi * 8 + r, r1 = 16 + lhi * 8 + r;
      float v;
      v = a00[r] + b0v; sBuf[HOFF + r0 * XSTR + colb + lrow]      = f2bf(0.5f * v * (1.0f + erff(v * 0.70710678118654752f)));
      v = a01[r] + b1v; sBuf[HOFF + r0 * XSTR + colb + 16 + lrow] = f2bf(0.5f * v * (1.0f + erff(v * 0.70710678118654752f)));
      v = a10[r] + b0v; sBuf[HOFF + r1 * XSTR + colb + lrow]      = f2bf(0.5f * v * (1.0f + erff(v * 0.70710678118654752f)));
      v = a11[r] + b1v; sBuf[HOFF + r1 * XSTR + colb + 16 + lrow] = f2bf(0.5f * v * (1.0f + erff(v * 0.70710678118654752f)));
    }
  }
  __syncthreads();

  // ---- fc2 (3 outputs) + trilinear skip ----
  if (tid < MTILE * 3) {
    int pl = tid / 3, o = tid - pl * 3;
    const unsigned short* hrow = &sBuf[HOFF + pl * XSTR];
    float s = fc2_b[o];
    for (int c = 0; c < 256; ++c) s += bf2f(hrow[c]) * fc2_w[o * 256 + c];

    int p = pos0 + pl;
    int t = p / (HOUT * WOUT);
    int rem = p - t * (HOUT * WOUT);
    int h = rem / WOUT, w = rem - h * WOUT;
    float ct = -1.0f + (1.0f / TOUT) + (2.0f / TOUT) * t;
    float ch = -1.0f + (1.0f / HOUT) + (2.0f / HOUT) * h;
    float cw = -1.0f + (1.0f / WOUT) + (2.0f / WOUT) * w;

    float xt = fminf(fmaxf(((ct + 1.0f) * TIN - 1.0f) * 0.5f, 0.0f), (float)(TIN - 1));
    int t0i = (int)floorf(xt); float ftf = xt - floorf(xt); int t1i = min(t0i + 1, TIN - 1);
    float xh = fminf(fmaxf(((ch + 1.0f) * HIN - 1.0f) * 0.5f, 0.0f), (float)(HIN - 1));
    int h0i = (int)floorf(xh); float fhf = xh - floorf(xh); int h1i = min(h0i + 1, HIN - 1);
    float xw = fminf(fmaxf(((cw + 1.0f) * WIN - 1.0f) * 0.5f, 0.0f), (float)(WIN - 1));
    int w0i = (int)floorf(xw); float fwf = xw - floorf(xw); int w1i = min(w0i + 1, WIN - 1);

    const float* so = sc + o * NPOS_IN;
    auto S = [&](int a, int b, int c) -> float { return so[a * (HIN * WIN) + b * WIN + c]; };
    float v00 = S(t0i, h0i, w0i) * (1.0f - ftf) + S(t1i, h0i, w0i) * ftf;
    float v01 = S(t0i, h0i, w1i) * (1.0f - ftf) + S(t1i, h0i, w1i) * ftf;
    float v10 = S(t0i, h1i, w0i) * (1.0f - ftf) + S(t1i, h1i, w0i) * ftf;
    float v11 = S(t0i, h1i, w1i) * (1.0f - ftf) + S(t1i, h1i, w1i) * ftf;
    float v0 = v00 * (1.0f - fhf) + v10 * fhf;
    float v1 = v01 * (1.0f - fhf) + v11 * fhf;
    out[o * NPOS + p] = s + (v0 * (1.0f - fwf) + v1 * fwf);
  }
}

extern "C" void kernel_launch(void* const* d_in, const int* in_sizes, int n_in,
                              void* d_out, int out_size, void* d_ws, size_t ws_size,
                              hipStream_t stream) {
  (void)in_sizes; (void)n_in; (void)out_size; (void)ws_size;
  const float* feat    = (const float*)d_in[0];
  const float* dw1_w   = (const float*)d_in[1];
  const float* dw1_b   = (const float*)d_in[2];
  const float* pw1_w   = (const float*)d_in[3];
  const float* pw1_b   = (const float*)d_in[4];
  const float* fc1_w   = (const float*)d_in[5];
  const float* fc1_b   = (const float*)d_in[6];
  const float* fc2_w   = (const float*)d_in[7];
  const float* fc2_b   = (const float*)d_in[8];
  const float* sc_dw_w = (const float*)d_in[9];
  const float* sc_dw_b = (const float*)d_in[10];
  const float* sc_pw_w = (const float*)d_in[11];
  const float* sc_pw_b = (const float*)d_in[12];
  float* out = (float*)d_out;

  char* ws = (char*)d_ws;
  unsigned short* w1s = (unsigned short*)(ws + W1S_OFF);
  unsigned short* w2s = (unsigned short*)(ws + W2S_OFF);
  float*          b1  = (float*)(ws + B1_OFF);
  float*          sc  = (float*)(ws + SC_OFF);

  k_fold_w1<<<544, 256, 0, stream>>>(pw1_w, dw1_w, w1s);
  k_fold_w2<<<256, 256, 0, stream>>>(fc1_w, w2s);
  k_b1<<<1, 256, 0, stream>>>(pw1_w, pw1_b, dw1_w, dw1_b, b1);
  k_sc<<<(NPOS_IN + 255) / 256, 256, 0, stream>>>(feat, sc_dw_w, sc_dw_b, sc_pw_w, sc_pw_b, sc);
  k_main<<<NPOS / MTILE, 256, 0, stream>>>(feat, w1s, w2s, b1, fc1_b, fc2_w, fc2_b, sc, out);
}
